// TransxnetHybridTokenMixer_39968965656586
// MI455X (gfx1250) — matrix operand order repaired and checked by reading, not compiled, not run on hardware
//
#include <hip/hip_runtime.h>
#include <cstdint>

typedef __attribute__((ext_vector_type(16))) __bf16 v16bf;
typedef __attribute__((ext_vector_type(8)))  float  v8f;

#define B_     8
#define DIM_   256
#define C1_    128
#define N_     3600
#define NK_    900
#define HEADS_ 4
#define INNER_ 32

// native f32 -> bf16 convert (RTNE in hardware on gfx1250)
__device__ __forceinline__ __bf16 f2bf(float f) { return (__bf16)f; }
__device__ __forceinline__ float geluf(float x) {
  return 0.5f * x * (1.0f + erff(x * 0.7071067811865475f));
}

// ---------------------------------------------------------------------------
// 1) 3x3 adaptive avg pool (20x20 regions) + global avg, per (b,c) plane
// ---------------------------------------------------------------------------
__global__ __launch_bounds__(128) void pool_k(const float* __restrict__ x,
                                              float* __restrict__ pooled,
                                              float* __restrict__ gavg) {
  __shared__ float sdata[9][128];
  int bc = blockIdx.x;               // b*128 + c, c in [0,128)
  int b = bc >> 7, c = bc & 127;
  const float* ip = x + ((long)b * DIM_ + c) * N_;
  float loc[9] = {0,0,0,0,0,0,0,0,0};
  for (int px = threadIdx.x; px < N_; px += 128) {
    int y = px / 60, xw = px % 60;
    loc[(y / 20) * 3 + (xw / 20)] += ip[px];
  }
  for (int r = 0; r < 9; r++) sdata[r][threadIdx.x] = loc[r];
  __syncthreads();
  for (int s = 64; s > 0; s >>= 1) {
    if (threadIdx.x < (unsigned)s)
      for (int r = 0; r < 9; r++) sdata[r][threadIdx.x] += sdata[r][threadIdx.x + s];
    __syncthreads();
  }
  if (threadIdx.x == 0) {
    float tot = 0.f;
    for (int r = 0; r < 9; r++) { float v = sdata[r][0]; pooled[(long)bc * 9 + r] = v * (1.f / 400.f); tot += v; }
    gavg[bc] = tot * (1.f / 3600.f);
  }
}

// ---------------------------------------------------------------------------
// 2) dynamic-conv projection MLP + softmax over G, produce per-(b,c) 3x3
//    kernels (wdyn) and biases (bdyn).  One block per batch.
// ---------------------------------------------------------------------------
__global__ __launch_bounds__(128) void dynmlp_k(
    const float* __restrict__ pooled, const float* __restrict__ gavg,
    const float* __restrict__ p1w, const float* __restrict__ p1b,
    const float* __restrict__ bns, const float* __restrict__ bnsh,
    const float* __restrict__ p2w, const float* __restrict__ p2b,
    const float* __restrict__ dynw, const float* __restrict__ dynb,
    float* __restrict__ wdyn, float* __restrict__ bdyn) {
  int b = blockIdx.x, tid = threadIdx.x;
  __shared__ float hid[10][32];
  __shared__ float outp[10][256];
  for (int t = tid; t < 320; t += 128) {        // hidden: 10 positions x 32 units
    int pos = t / 32, u = t % 32;
    float acc = p1b[u];
    for (int c = 0; c < 128; c++) {
      float xv = (pos < 9) ? pooled[((long)b * 128 + c) * 9 + pos] : gavg[b * 128 + c];
      acc += p1w[u * 128 + c] * xv;
    }
    hid[pos][u] = geluf(acc * bns[u] + bnsh[u]);
  }
  __syncthreads();
  for (int t = tid; t < 2560; t += 128) {       // out: 10 positions x 256
    int pos = t / 256, o = t % 256;
    float acc = p2b[o];
    for (int u = 0; u < 32; u++) acc += p2w[o * 32 + u] * hid[pos][u];
    outp[pos][o] = acc;
  }
  __syncthreads();
  if (tid < 128) {
    int c = tid;
    for (int r = 0; r < 9; r++) {
      float a0 = outp[r][c], a1 = outp[r][128 + c];
      float mx = fmaxf(a0, a1);
      float e0 = expf(a0 - mx), e1 = expf(a1 - mx), inv = 1.f / (e0 + e1);
      wdyn[((long)b * 128 + c) * 9 + r] =
          (e0 * dynw[c * 9 + r] + e1 * dynw[1152 + c * 9 + r]) * inv;
    }
    float a0 = outp[9][c], a1 = outp[9][128 + c];
    float mx = fmaxf(a0, a1);
    float e0 = expf(a0 - mx), e1 = expf(a1 - mx), inv = 1.f / (e0 + e1);
    bdyn[b * 128 + c] = (e0 * dynb[c] + e1 * dynb[128 + c]) * inv;
  }
}

// ---------------------------------------------------------------------------
// 3) dynamic depthwise 3x3 conv on x1, writes y1 into xc[:, :128]
// ---------------------------------------------------------------------------
__global__ __launch_bounds__(256) void dyndw_k(const float* __restrict__ x,
                                               const float* __restrict__ wdyn,
                                               const float* __restrict__ bdyn,
                                               float* __restrict__ xc) {
  int bc = blockIdx.x;
  int b = bc >> 7, c = bc & 127;
  const float* ip = x + ((long)b * DIM_ + c) * N_;
  float wk[9];
#pragma unroll
  for (int i = 0; i < 9; i++) wk[i] = wdyn[(long)bc * 9 + i];
  float bs = bdyn[bc];
  float* op = xc + ((long)b * DIM_ + c) * N_;
  for (int px = threadIdx.x; px < N_; px += blockDim.x) {
    int oy = px / 60, ox = px % 60;
    float acc = bs;
#pragma unroll
    for (int ky = 0; ky < 3; ky++) {
      int iy = oy - 1 + ky; if ((unsigned)iy >= 60u) continue;
#pragma unroll
      for (int kx = 0; kx < 3; kx++) {
        int ix = ox - 1 + kx; if ((unsigned)ix >= 60u) continue;
        acc += ip[iy * 60 + ix] * wk[ky * 3 + kx];
      }
    }
    op[px] = acc;
  }
}

// ---------------------------------------------------------------------------
// 4) generic depthwise conv. EPI: 0 none, 1 gelu(bn(x)), 2 bn(gelu(x)), 4 +res
// ---------------------------------------------------------------------------
template <int KS, int STRIDE, int EPI>
__global__ __launch_bounds__(256) void dwconv_k(
    const float* __restrict__ in, long in_bstride,
    const float* __restrict__ w, const float* __restrict__ bias,
    const float* __restrict__ bns, const float* __restrict__ bnsh,
    const float* __restrict__ res, long res_bstride,
    float* __restrict__ out, long out_bstride,
    int C, int Hin, int Win, int Ho, int Wo, int pad) {
  int bc = blockIdx.x;
  int b = bc / C, c = bc % C;
  const float* ip = in + (long)b * in_bstride + (long)c * Hin * Win;
  const float* wp = w + c * KS * KS;
  float bs = bias[c];
  for (int px = threadIdx.x; px < Ho * Wo; px += blockDim.x) {
    int oy = px / Wo, ox = px % Wo;
    float acc = bs;
#pragma unroll
    for (int ky = 0; ky < KS; ky++) {
      int iy = oy * STRIDE - pad + ky;
      if ((unsigned)iy >= (unsigned)Hin) continue;
#pragma unroll
      for (int kx = 0; kx < KS; kx++) {
        int ix = ox * STRIDE - pad + kx;
        if ((unsigned)ix >= (unsigned)Win) continue;
        acc += ip[iy * Win + ix] * wp[ky * KS + kx];
      }
    }
    float v = acc;
    if (EPI == 1) v = geluf(v * bns[c] + bnsh[c]);
    else if (EPI == 2) v = geluf(v) * bns[c] + bnsh[c];
    else if (EPI == 4) v = v + res[(long)b * res_bstride + (long)c * Ho * Wo + px];
    out[(long)b * out_bstride + (long)c * Ho * Wo + px] = v;
  }
}

// ---------------------------------------------------------------------------
// 5) 1x1 conv = GEMM: out[b,o,n] = sum_c W[o,c]*in[b,c,n] (+epilogue), WMMA.
//    One wave -> one 16x16 output tile, K-loop step 32 (bf16, f32 accum).
//    EPI: 0 +bias, 2 bn(gelu(x+bias)), 3 (x+bias)*s+sh + residual
// ---------------------------------------------------------------------------
template <int EPI>
__global__ __launch_bounds__(128) void gemm1x1_wmma(
    const float* __restrict__ in, long in_bstride,
    const float* __restrict__ W, const float* __restrict__ bias,
    const float* __restrict__ bns, const float* __restrict__ bnsh,
    const float* __restrict__ res, long res_bstride,
    float* __restrict__ out, long out_bstride,
    int C, int N, int ntiles) {
  int wave = threadIdx.x >> 5;
  int lane = threadIdx.x & 31;
  int half = lane >> 4, m = lane & 15;
  int nt = blockIdx.x * 4 + wave;
  if (nt >= ntiles) nt = ntiles - 1;
  int n0 = nt * 16; if (n0 > N - 16) n0 = N - 16;
  int o0 = blockIdx.y * 16;
  int b = blockIdx.z;
  const float* inb = in + (long)b * in_bstride;
  v8f acc = {0.f, 0.f, 0.f, 0.f, 0.f, 0.f, 0.f, 0.f};
  for (int c0 = 0; c0 < C; c0 += 32) {
    v16bf A, Bm;
#pragma unroll
    for (int j = 0; j < 8; j++) {
      int ka = (j < 4 ? 2 * j : 16 + 2 * (j - 4)) + half * 8;   // A 16x32 layout
      A[2 * j]     = f2bf(W[(o0 + m) * C + c0 + ka]);
      A[2 * j + 1] = f2bf(W[(o0 + m) * C + c0 + ka + 1]);
      int kb = (j < 4 ? 2 * j : 16 + 2 * (j - 4)) + half * 8;   // B 32x16 layout
      Bm[2 * j]     = f2bf(inb[(long)(c0 + kb) * N + n0 + m]);
      Bm[2 * j + 1] = f2bf(inb[(long)(c0 + kb + 1) * N + n0 + m]);
    }
    acc = __builtin_amdgcn_wmma_f32_16x16x32_bf16(false, A, false, Bm,
                                                  (short)0, acc, false, false);
  }
#pragma unroll
  for (int r = 0; r < 8; r++) {
    int o = o0 + r + half * 8;
    int n = n0 + m;
    float v = acc[r] + bias[o];
    if (EPI == 2) v = geluf(v) * bns[o] + bnsh[o];
    else if (EPI == 3) v = v * bns[o] + bnsh[o] +
                           res[(long)b * res_bstride + (long)o * N + n];
    out[(long)b * out_bstride + (long)o * N + n] = v;
  }
}

// ---------------------------------------------------------------------------
// 6) fused OSRA attention (flash-style, online softmax), WMMA for QK^T and PV.
//    One wave = 16 query rows of one (b,h). 3600 = 225*16 row tiles exactly.
//    Nk = 900 processed in 29 chunks of 32 columns (tail masked).
//    Writes y2 into xc[:, 128:].
// ---------------------------------------------------------------------------
__global__ __launch_bounds__(128) void osra_attn_k(
    const float* __restrict__ qbuf, const float* __restrict__ kvo,
    const float* __restrict__ rel, float* __restrict__ xc) {
  __shared__ __bf16 Plds[4][16][34];
  int wave = threadIdx.x >> 5, lane = threadIdx.x & 31;
  int half = lane >> 4, m = lane & 15;
  int h = blockIdx.y, b = blockIdx.z;
  int rt = blockIdx.x * 4 + wave; if (rt > 224) rt = 224;
  int r0 = rt * 16;
  const float* qb = qbuf + (long)b * C1_ * N_ + (long)(h * 32) * N_;
  const float* kb = kvo + (long)b * DIM_ * NK_ + (long)(h * 32) * NK_;
  const float* vb = kvo + (long)b * DIM_ * NK_ + (long)(128 + h * 32) * NK_;
  const float* rb = rel + (long)h * N_ * NK_;

  v16bf Aq;
#pragma unroll
  for (int j = 0; j < 8; j++) {
    int ka = (j < 4 ? 2 * j : 16 + 2 * (j - 4)) + half * 8;
    Aq[2 * j]     = f2bf(qb[(long)ka * N_ + r0 + m]);
    Aq[2 * j + 1] = f2bf(qb[(long)(ka + 1) * N_ + r0 + m]);
  }
  float rmax[8], lsum[8];
  v8f O0 = {0.f,0.f,0.f,0.f,0.f,0.f,0.f,0.f};
  v8f O1 = {0.f,0.f,0.f,0.f,0.f,0.f,0.f,0.f};
#pragma unroll
  for (int r = 0; r < 8; r++) { rmax[r] = -1e30f; lsum[r] = 0.f; }

  const float scale = 0.17677669529663689f;   // 32^-0.5
  for (int cidx = 0; cidx < 29; ++cidx) {
    int nk0 = cidx * 32;
    if (cidx < 28) __builtin_prefetch(&rb[(long)(r0 + half * 8) * NK_ + nk0 + 32 + m], 0, 1);
    v8f S[2];
#pragma unroll
    for (int t = 0; t < 2; t++) {
      int col = nk0 + t * 16 + m;
      int cc = col < NK_ ? col : NK_ - 1;
      v16bf Bk;
#pragma unroll
      for (int j = 0; j < 8; j++) {
        int kk = (j < 4 ? 2 * j : 16 + 2 * (j - 4)) + half * 8;
        float v0 = kb[(long)kk * NK_ + cc];
        float v1 = kb[(long)(kk + 1) * NK_ + cc];
        Bk[2 * j]     = f2bf(col < NK_ ? v0 : 0.f);
        Bk[2 * j + 1] = f2bf(col < NK_ ? v1 : 0.f);
      }
      v8f z = {0.f,0.f,0.f,0.f,0.f,0.f,0.f,0.f};
      S[t] = __builtin_amdgcn_wmma_f32_16x16x32_bf16(false, Aq, false, Bk,
                                                     (short)0, z, false, false);
#pragma unroll
      for (int r = 0; r < 8; r++) {
        int row = r0 + r + half * 8;
        float sv = S[t][r] * scale;
        sv += (col < NK_) ? rb[(long)row * NK_ + col] : -1e30f;
        S[t][r] = sv;
      }
    }
    // online softmax (row stats across the 16 lanes of each half-wave)
    float fac[8];
#pragma unroll
    for (int r = 0; r < 8; r++) {
      float cm = fmaxf(S[0][r], S[1][r]);
      for (int msk = 8; msk >= 1; msk >>= 1) cm = fmaxf(cm, __shfl_xor(cm, msk, 32));
      float nm = fmaxf(rmax[r], cm);
      float f = expf(rmax[r] - nm);
      float p0 = expf(S[0][r] - nm), p1 = expf(S[1][r] - nm);
      float rs = p0 + p1;
      for (int msk = 8; msk >= 1; msk >>= 1) rs += __shfl_xor(rs, msk, 32);
      lsum[r] = lsum[r] * f + rs;
      rmax[r] = nm; fac[r] = f;
      S[0][r] = p0; S[1][r] = p1;
    }
#pragma unroll
    for (int r = 0; r < 8; r++) { O0[r] *= fac[r]; O1[r] *= fac[r]; }

    // transpose P (C-layout) -> A-fragment layout through LDS
    __syncthreads();
#pragma unroll
    for (int r = 0; r < 8; r++) {
      Plds[wave][r + half * 8][m]      = f2bf(S[0][r]);
      Plds[wave][r + half * 8][16 + m] = f2bf(S[1][r]);
    }
    __syncthreads();
    v16bf AP;
#pragma unroll
    for (int j = 0; j < 8; j++) {
      int ka = (j < 4 ? 2 * j : 16 + 2 * (j - 4)) + half * 8;
      AP[2 * j]     = Plds[wave][m][ka];
      AP[2 * j + 1] = Plds[wave][m][ka + 1];
    }
    // P @ V  (two 16-wide d tiles)
#pragma unroll
    for (int dt = 0; dt < 2; dt++) {
      v16bf BV;
#pragma unroll
      for (int j = 0; j < 8; j++) {
        int kk = (j < 4 ? 2 * j : 16 + 2 * (j - 4)) + half * 8;
        int n0a = nk0 + kk;     int c0a = n0a < NK_ ? n0a : NK_ - 1;
        int n1a = nk0 + kk + 1; int c1a = n1a < NK_ ? n1a : NK_ - 1;
        BV[2 * j]     = f2bf(vb[(long)(dt * 16 + m) * NK_ + c0a]);
        BV[2 * j + 1] = f2bf(vb[(long)(dt * 16 + m) * NK_ + c1a]);
      }
      if (dt == 0)
        O0 = __builtin_amdgcn_wmma_f32_16x16x32_bf16(false, AP, false, BV,
                                                     (short)0, O0, false, false);
      else
        O1 = __builtin_amdgcn_wmma_f32_16x16x32_bf16(false, AP, false, BV,
                                                     (short)0, O1, false, false);
    }
  }
  // normalize and store y2 into xc channels [128, 256)
#pragma unroll
  for (int r = 0; r < 8; r++) {
    float inv = 1.f / lsum[r];
    int token = r0 + r + half * 8;
    int ch0 = 128 + h * 32 + m;
    xc[((long)b * DIM_ + ch0) * N_ + token]      = O0[r] * inv;
    xc[((long)b * DIM_ + ch0 + 16) * N_ + token] = O1[r] * inv;
  }
}

// ---------------------------------------------------------------------------
extern "C" void kernel_launch(void* const* d_in, const int* in_sizes, int n_in,
                              void* d_out, int out_size, void* d_ws, size_t ws_size,
                              hipStream_t stream) {
  const float* x        = (const float*)d_in[0];
  const float* rel_pos  = (const float*)d_in[1];
  const float* dyn_w    = (const float*)d_in[2];
  const float* dyn_b    = (const float*)d_in[3];
  const float* dyn_p1_w = (const float*)d_in[4];
  const float* dyn_p1_b = (const float*)d_in[5];
  const float* dyn_bn_s = (const float*)d_in[6];
  const float* dyn_bn_sh= (const float*)d_in[7];
  const float* dyn_p2_w = (const float*)d_in[8];
  const float* dyn_p2_b = (const float*)d_in[9];
  const float* wq       = (const float*)d_in[10];
  const float* bq       = (const float*)d_in[11];
  const float* sr_dw_w  = (const float*)d_in[12];
  const float* sr_dw_b  = (const float*)d_in[13];
  const float* sr_bn_s  = (const float*)d_in[14];
  const float* sr_bn_sh = (const float*)d_in[15];
  const float* sr_pw_w  = (const float*)d_in[16];
  const float* sr_pw_b  = (const float*)d_in[17];
  const float* lc_w     = (const float*)d_in[18];
  const float* lc_b     = (const float*)d_in[19];
  const float* wkv      = (const float*)d_in[20];
  const float* bkv      = (const float*)d_in[21];
  const float* pj_dw_w  = (const float*)d_in[22];
  const float* pj_dw_b  = (const float*)d_in[23];
  const float* pj_bn1_s = (const float*)d_in[24];
  const float* pj_bn1_sh= (const float*)d_in[25];
  const float* pj_pw1_w = (const float*)d_in[26];
  const float* pj_pw1_b = (const float*)d_in[27];
  const float* pj_bn2_s = (const float*)d_in[28];
  const float* pj_bn2_sh= (const float*)d_in[29];
  const float* pj_pw2_w = (const float*)d_in[30];
  const float* pj_pw2_b = (const float*)d_in[31];
  const float* pj_bn3_s = (const float*)d_in[32];
  const float* pj_bn3_sh= (const float*)d_in[33];
  float* out = (float*)d_out;

  // workspace carve-up (floats)
  float* ws = (float*)d_ws;
  size_t o = 0;
  float* xc  = ws + o; o += (size_t)B_ * DIM_ * N_;     // y1|y2 concat
  float* q   = ws + o; o += (size_t)B_ * C1_ * N_;
  float* kvd = ws + o; o += (size_t)B_ * C1_ * NK_;     // sr dw out (post bn+gelu)
  float* kvp = ws + o; o += (size_t)B_ * C1_ * NK_;     // after sr_pw
  float* kv2 = ws + o; o += (size_t)B_ * C1_ * NK_;     // after local conv + res
  float* kvo = ws + o; o += (size_t)B_ * DIM_ * NK_;    // k|v
  float* t1  = ws + o; o += (size_t)B_ * DIM_ * N_;     // pj dw out (post gelu+bn)
  float* t2  = ws + o; o += (size_t)B_ * INNER_ * N_;   // pw1 out (post gelu+bn)
  float* pooled = ws + o; o += (size_t)B_ * C1_ * 9;
  float* gavg   = ws + o; o += (size_t)B_ * C1_;
  float* wdyn   = ws + o; o += (size_t)B_ * C1_ * 9;
  float* bdyn   = ws + o; o += (size_t)B_ * C1_;

  // ---- dynamic conv branch (writes xc[:, :128]) ----
  pool_k<<<B_ * C1_, 128, 0, stream>>>(x, pooled, gavg);
  dynmlp_k<<<B_, 128, 0, stream>>>(pooled, gavg, dyn_p1_w, dyn_p1_b, dyn_bn_s,
                                   dyn_bn_sh, dyn_p2_w, dyn_p2_b, dyn_w, dyn_b,
                                   wdyn, bdyn);
  dyndw_k<<<B_ * C1_, 256, 0, stream>>>(x, wdyn, bdyn, xc);

  // ---- attention branch ----
  // q = conv1x1(x2)
  gemm1x1_wmma<0><<<dim3(57, 8, B_), 128, 0, stream>>>(
      x + (long)C1_ * N_, (long)DIM_ * N_, wq, bq, nullptr, nullptr, nullptr, 0,
      q, (long)C1_ * N_, C1_, N_, 225);
  // sr: dw5x5 s2 then gelu(bn(.))
  dwconv_k<5, 2, 1><<<B_ * C1_, 256, 0, stream>>>(
      x + (long)C1_ * N_, (long)DIM_ * N_, sr_dw_w, sr_dw_b, sr_bn_s, sr_bn_sh,
      nullptr, 0, kvd, (long)C1_ * NK_, C1_, 60, 60, 30, 30, 2);
  // sr pointwise
  gemm1x1_wmma<0><<<dim3(15, 8, B_), 128, 0, stream>>>(
      kvd, (long)C1_ * NK_, sr_pw_w, sr_pw_b, nullptr, nullptr, nullptr, 0,
      kvp, (long)C1_ * NK_, C1_, NK_, 57);
  // local conv + residual
  dwconv_k<3, 1, 4><<<B_ * C1_, 256, 0, stream>>>(
      kvp, (long)C1_ * NK_, lc_w, lc_b, nullptr, nullptr, kvp, (long)C1_ * NK_,
      kv2, (long)C1_ * NK_, C1_, 30, 30, 30, 30, 1);
  // kv projection (k|v)
  gemm1x1_wmma<0><<<dim3(15, 16, B_), 128, 0, stream>>>(
      kv2, (long)C1_ * NK_, wkv, bkv, nullptr, nullptr, nullptr, 0,
      kvo, (long)DIM_ * NK_, C1_, NK_, 57);
  // fused attention -> xc[:, 128:]
  osra_attn_k<<<dim3(57, HEADS_, B_), 128, 0, stream>>>(q, kvo, rel_pos, xc);

  // ---- fuse + proj + residual ----
  dwconv_k<3, 1, 2><<<B_ * DIM_, 256, 0, stream>>>(
      xc, (long)DIM_ * N_, pj_dw_w, pj_dw_b, pj_bn1_s, pj_bn1_sh, nullptr, 0,
      t1, (long)DIM_ * N_, DIM_, 60, 60, 60, 60, 1);
  gemm1x1_wmma<2><<<dim3(57, 2, B_), 128, 0, stream>>>(
      t1, (long)DIM_ * N_, pj_pw1_w, pj_pw1_b, pj_bn2_s, pj_bn2_sh, nullptr, 0,
      t2, (long)INNER_ * N_, DIM_, N_, 225);
  gemm1x1_wmma<3><<<dim3(57, 16, B_), 128, 0, stream>>>(
      t2, (long)INNER_ * N_, pj_pw2_w, pj_pw2_b, pj_bn3_s, pj_bn3_sh,
      xc, (long)DIM_ * N_, out, (long)DIM_ * N_, INNER_, N_, 225);
}
